// MultiHeadAttentionWithRoPE_66185446031511
// MI455X (gfx1250) — compile-verified, matrix operand order checked by reading, not run on hardware
//
#include <hip/hip_runtime.h>
#include <hip/hip_bf16.h>

// MHA + RoPE for MI455X (gfx1250, wave32, WMMA).
// v2.1: register-blocked WMMA GEMM (64x64/wave -> 1 b128 load per wmma),
//       attention with 32 queries/wave (2x K/V fragment reuse) and
//       double-buffered async global->LDS staging of K/V tiles (ASYNCcnt path).
//       Async builtin takes (v4i addrspace(1)*, v4i addrspace(3)*, Ii, Ii).

typedef __attribute__((ext_vector_type(16))) _Float16 v16h;
typedef __attribute__((ext_vector_type(8)))  float    v8f;
typedef int v4i_vec __attribute__((vector_size(16)));

#define BATCH 4
#define SEQ   2048
#define DMODEL 1024
#define NHEAD 16
#define HDIM  64

#if defined(__has_builtin)
#if __has_builtin(__builtin_amdgcn_global_load_async_to_lds_b128) && \
    __has_builtin(__builtin_amdgcn_s_wait_asynccnt)
#define HAVE_ASYNC_LDS 1
#endif
#endif

union H8 { float4 f4; _Float16 h[8]; };

// Load a 16-half WMMA fragment (A or B operand) for this lane.
// Layout (ISA 7.12.2, 16-bit A 16x32): halves h<8 -> K = k0 + hi*8 + h,
// halves h>=8 -> K = k0 + 16 + hi*8 + (h-8).  Two contiguous 16B chunks.
__device__ inline v16h load_frag16(const _Float16* __restrict__ row_ptr, int k0, int hi) {
    H8 g0, g1;
    g0.f4 = *(const float4*)(row_ptr + k0 + hi * 8);
    g1.f4 = *(const float4*)(row_ptr + k0 + 16 + hi * 8);
    v16h r;
#pragma unroll
    for (int i = 0; i < 8; ++i) { r[i] = g0.h[i]; r[i + 8] = g1.h[i]; }
    return r;
}

__device__ inline v8f wmma_f16(v16h a, v16h b, v8f c) {
    return __builtin_amdgcn_wmma_f32_16x16x32_f16(false, a, false, b, (short)0, c, false, false);
}

// 16-byte global -> LDS copy: async (ASYNCcnt) when available, else through VGPRs.
__device__ inline void cp16_g2l(const _Float16* g, _Float16* l) {
#ifdef HAVE_ASYNC_LDS
    __builtin_amdgcn_global_load_async_to_lds_b128(
        (__attribute__((address_space(1))) v4i_vec*)(g),
        (__attribute__((address_space(3))) v4i_vec*)(l), 0, 0);
#else
    *(float4*)l = *(const float4*)g;
#endif
}

// Reductions across the 16 lanes of each wave-half (matches C-matrix rows).
__device__ inline float half_max16(float x) {
    x = fmaxf(x, __shfl_xor(x, 1, 32));
    x = fmaxf(x, __shfl_xor(x, 2, 32));
    x = fmaxf(x, __shfl_xor(x, 4, 32));
    x = fmaxf(x, __shfl_xor(x, 8, 32));
    return x;
}
__device__ inline float half_sum16(float x) {
    x += __shfl_xor(x, 1, 32);
    x += __shfl_xor(x, 2, 32);
    x += __shfl_xor(x, 4, 32);
    x += __shfl_xor(x, 8, 32);
    return x;
}

// ---------------- f32 -> f16 conversion ----------------
__global__ void cvt_f16_kernel(const float* __restrict__ in, _Float16* __restrict__ out, int n) {
    int i = blockIdx.x * blockDim.x + threadIdx.x;
    if (i < n) out[i] = (_Float16)in[i];
}

// ---------------- WMMA GEMM: out = A(MxK) @ Bt(NxK)^T + bias ----------------
// 64x64 tile per wave = 4x4 WMMA tiles: 16 wmma per k-step for 16 b128 loads.
// mode 0/1: store f16 head-major  [bh][s][d]
// mode 2  : store f16 transposed  [bh][d][s]   (V, so PV B-operand is k-contiguous)
// mode 3  : store f32 flat        [m][n]       (final projection)
__global__ void __launch_bounds__(256)
gemm_wmma_kernel(const _Float16* __restrict__ A, const _Float16* __restrict__ Bt,
                 const float* __restrict__ bias, void* __restrict__ out, int mode) {
    const int M = BATCH * SEQ, N = DMODEL, K = DMODEL;
    int wave = blockIdx.x * (blockDim.x >> 5) + (threadIdx.x >> 5);
    const int ntiles = (M / 64) * (N / 64);
    if (wave >= ntiles) return;
    int mt = wave / (N / 64);
    int nt = wave % (N / 64);
    int lane = threadIdx.x & 31, nlo = lane & 15, hi = lane >> 4;

    const _Float16* arows[4];
    const _Float16* brows[4];
#pragma unroll
    for (int i = 0; i < 4; ++i) {
        arows[i] = A + (size_t)(mt * 64 + i * 16 + nlo) * K;   // A-frag rows (M = lane&15)
        brows[i] = Bt + (size_t)(nt * 64 + i * 16 + nlo) * K;  // B-frag rows (N = lane&15)
    }

    v8f acc[4][4];
#pragma unroll
    for (int i = 0; i < 4; ++i)
#pragma unroll
        for (int j = 0; j < 4; ++j) acc[i][j] = (v8f){};

    for (int k0 = 0; k0 < K; k0 += 32) {
        v16h af[4], bf[4];
#pragma unroll
        for (int i = 0; i < 4; ++i) af[i] = load_frag16(arows[i], k0, hi);
#pragma unroll
        for (int j = 0; j < 4; ++j) bf[j] = load_frag16(brows[j], k0, hi);
#pragma unroll
        for (int i = 0; i < 4; ++i)
#pragma unroll
            for (int j = 0; j < 4; ++j)
                acc[i][j] = wmma_f16(af[i], bf[j], acc[i][j]);
    }

#pragma unroll
    for (int j = 0; j < 4; ++j) {
        int n = nt * 64 + j * 16 + nlo;
        float bv = bias[n];
        if (mode == 3) {
            float* o = (float*)out;
#pragma unroll
            for (int i = 0; i < 4; ++i)
#pragma unroll
                for (int v = 0; v < 8; ++v) {
                    int m = mt * 64 + i * 16 + v + 8 * hi;     // C-layout row
                    o[(size_t)m * N + n] = acc[i][j][v] + bv;
                }
        } else {
            _Float16* o = (_Float16*)out;
            int h = n >> 6, d = n & 63;
#pragma unroll
            for (int i = 0; i < 4; ++i)
#pragma unroll
                for (int v = 0; v < 8; ++v) {
                    int m = mt * 64 + i * 16 + v + 8 * hi;
                    int b_ = m >> 11;                          // m / SEQ
                    int s  = m & (SEQ - 1);
                    int bh = b_ * NHEAD + h;
                    size_t idx = (mode == 2)
                        ? ((size_t)(bh * HDIM + d) * SEQ + s)  // V transposed
                        : ((size_t)(bh * SEQ + s) * HDIM + d); // Q/K head-major
                    o[idx] = (_Float16)(acc[i][j][v] + bv);
                }
        }
    }
}

// ---------------- RoPE in-place on f16 Q and K ----------------
__global__ void rope_kernel(_Float16* __restrict__ Qh, _Float16* __restrict__ Kh) {
    const int total = BATCH * NHEAD * SEQ * (HDIM / 2);
    int idx = blockIdx.x * blockDim.x + threadIdx.x;
    if (idx >= total) return;
    int d  = idx & 31;
    int s  = (idx >> 5) & (SEQ - 1);
    int bh = idx >> 16;
    float inv = __expf(-(2.0f * d / 64.0f) * 9.210340371976184f);  // 10000^(-2d/64)
    float fr  = (float)s * inv;
    float cs  = cosf(fr), sn = sinf(fr);
    size_t base = (size_t)(bh * SEQ + s) * HDIM;
    float q1 = (float)Qh[base + d], q2 = (float)Qh[base + d + 32];
    Qh[base + d]      = (_Float16)(q1 * cs - q2 * sn);
    Qh[base + d + 32] = (_Float16)(q1 * sn + q2 * cs);
    float k1 = (float)Kh[base + d], k2 = (float)Kh[base + d + 32];
    Kh[base + d]      = (_Float16)(k1 * cs - k2 * sn);
    Kh[base + d + 32] = (_Float16)(k1 * sn + k2 * cs);
}

// ---------------- Flash attention: 1 wave per (bh, 32-query tile) ----------------
// K/V 32-key tiles double-buffered in LDS via async global->LDS copies.
__global__ void __launch_bounds__(32)
attn_kernel(const _Float16* __restrict__ Qh, const _Float16* __restrict__ Kh,
            const _Float16* __restrict__ Vt, _Float16* __restrict__ AO) {
    int qt = blockIdx.x;          // 32-query tile
    int bh = blockIdx.y;          // batch*head
    int lane = threadIdx.x & 31, nlo = lane & 15, hi = lane >> 4;

    __shared__ alignas(16) _Float16 Kst[2][32 * HDIM];   // [key][d]
    __shared__ alignas(16) _Float16 Vst[2][HDIM * 32];   // [d][key]
    __shared__ alignas(16) _Float16 pbuf[32 * 32];       // P-tile transpose staging

    const _Float16* kbase = Kh + (size_t)bh * SEQ * HDIM;
    const _Float16* vbase = Vt + (size_t)bh * HDIM * SEQ;

    // Q fragments: 2 q-subtiles x 2 k-halves
    v16h aq[2][2];
#pragma unroll
    for (int q = 0; q < 2; ++q) {
        const _Float16* qrow = Qh + (size_t)(bh * SEQ + qt * 32 + q * 16 + nlo) * HDIM;
        aq[q][0] = load_frag16(qrow, 0, hi);
        aq[q][1] = load_frag16(qrow, 32, hi);
    }

    v8f o[2][4];
    float mrun[2][8], lrun[2][8];
#pragma unroll
    for (int q = 0; q < 2; ++q) {
#pragma unroll
        for (int t = 0; t < 4; ++t) o[q][t] = (v8f){};
#pragma unroll
        for (int v = 0; v < 8; ++v) { mrun[q][v] = -1e30f; lrun[q][v] = 0.0f; }
    }

    // Stage (K,V) tile for key block kb into buffer b.
    auto stage = [&](int kb, int b) {
        // K tile: rows kb..kb+31 are 4096 contiguous bytes in Kh.
        const _Float16* kg = kbase + (size_t)kb * HDIM;
#pragma unroll
        for (int c = 0; c < 8; ++c)
            cp16_g2l(kg + (c * 32 + lane) * 8, &Kst[b][(c * 32 + lane) * 8]);
        // V tile: 64 rows (d) x 64B windows at column kb.
        const _Float16* vg = vbase + kb;
#pragma unroll
        for (int rr = 0; rr < 2; ++rr) {
            int row = rr * 32 + lane;
#pragma unroll
            for (int c = 0; c < 4; ++c)
                cp16_g2l(vg + (size_t)row * SEQ + c * 8, &Vst[b][row * 32 + c * 8]);
        }
    };

    stage(0, 0);
    int buf = 0;

    for (int kb = 0; kb < SEQ; kb += 32) {
        __syncthreads();
#ifdef HAVE_ASYNC_LDS
        __builtin_amdgcn_s_wait_asynccnt(0);
#endif
        if (kb + 32 < SEQ) stage(kb + 32, buf ^ 1);

        const _Float16* kt = &Kst[buf][0];
        const _Float16* vt = &Vst[buf][0];

        // S = Q @ K^T: B-fragment loaded once, used by both q-subtiles.
        v8f s[2][2];
#pragma unroll
        for (int q = 0; q < 2; ++q) { s[q][0] = (v8f){}; s[q][1] = (v8f){}; }
#pragma unroll
        for (int nt = 0; nt < 2; ++nt)
#pragma unroll
            for (int kk = 0; kk < 2; ++kk) {
                v16h bfr = load_frag16(kt + (nt * 16 + nlo) * HDIM, kk * 32, hi);
                s[0][nt] = wmma_f16(aq[0][kk], bfr, s[0][nt]);
                s[1][nt] = wmma_f16(aq[1][kk], bfr, s[1][nt]);
            }

        // Online softmax per q-subtile; row M = v + 8*hi lives in one 16-lane half.
#pragma unroll
        for (int q = 0; q < 2; ++q) {
            float alpha[8];
#pragma unroll
            for (int v = 0; v < 8; ++v) {
                float x0 = s[q][0][v] * 0.125f;        // 1/sqrt(64)
                float x1 = s[q][1][v] * 0.125f;
                float mx = half_max16(fmaxf(x0, x1));
                float mnew = fmaxf(mrun[q][v], mx);
                alpha[v] = __expf(mrun[q][v] - mnew);
                mrun[q][v] = mnew;
                float p0 = __expf(x0 - mnew);
                float p1 = __expf(x1 - mnew);
                s[q][0][v] = p0; s[q][1][v] = p1;
                lrun[q][v] = lrun[q][v] * alpha[v] + half_sum16(p0 + p1);
            }
#pragma unroll
            for (int v = 0; v < 8; ++v) {
                o[q][0][v] *= alpha[v]; o[q][1][v] *= alpha[v];
                o[q][2][v] *= alpha[v]; o[q][3][v] *= alpha[v];
            }
        }

        // Transpose P (C layout) -> A-fragment layout via LDS.
        __syncthreads();
#pragma unroll
        for (int q = 0; q < 2; ++q)
#pragma unroll
            for (int v = 0; v < 8; ++v) {
                int m = q * 16 + v + 8 * hi;
                pbuf[m * 32 + nlo]      = (_Float16)s[q][0][v];
                pbuf[m * 32 + 16 + nlo] = (_Float16)s[q][1][v];
            }
        __syncthreads();
        v16h ap[2];
        ap[0] = load_frag16(pbuf + nlo * 32, 0, hi);
        ap[1] = load_frag16(pbuf + (16 + nlo) * 32, 0, hi);

        // O += P @ V: V-fragment loaded once, used by both q-subtiles.
#pragma unroll
        for (int t = 0; t < 4; ++t) {
            v16h bv = load_frag16(vt + (t * 16 + nlo) * 32, 0, hi);
            o[0][t] = wmma_f16(ap[0], bv, o[0][t]);
            o[1][t] = wmma_f16(ap[1], bv, o[1][t]);
        }

        buf ^= 1;
    }

    // Epilogue: normalize and store f16 to AO[b][s][h*64+d]
    int b_ = bh >> 4, h = bh & 15;
#pragma unroll
    for (int q = 0; q < 2; ++q)
#pragma unroll
        for (int v = 0; v < 8; ++v) {
            int srow = qt * 32 + q * 16 + v + 8 * hi;
            float inv_l = 1.0f / lrun[q][v];
            size_t base = (size_t)(b_ * SEQ + srow) * DMODEL + h * HDIM;
            AO[base + nlo]      = (_Float16)(o[q][0][v] * inv_l);
            AO[base + 16 + nlo] = (_Float16)(o[q][1][v] * inv_l);
            AO[base + 32 + nlo] = (_Float16)(o[q][2][v] * inv_l);
            AO[base + 48 + nlo] = (_Float16)(o[q][3][v] * inv_l);
        }
}

extern "C" void kernel_launch(void* const* d_in, const int* in_sizes, int n_in,
                              void* d_out, int out_size, void* d_ws, size_t ws_size,
                              hipStream_t stream) {
    const float* query = (const float*)d_in[0];
    const float* key   = (const float*)d_in[1];
    const float* value = (const float*)d_in[2];
    const float* Wq = (const float*)d_in[3]; const float* bq = (const float*)d_in[4];
    const float* Wk = (const float*)d_in[5]; const float* bk = (const float*)d_in[6];
    const float* Wv = (const float*)d_in[7]; const float* bv = (const float*)d_in[8];
    const float* Wo = (const float*)d_in[9]; const float* bo = (const float*)d_in[10];
    float* out = (float*)d_out;

    const size_t NX = (size_t)BATCH * SEQ * DMODEL;   // 8388608
    const size_t NW = (size_t)DMODEL * DMODEL;        // 1048576

    _Float16* p   = (_Float16*)d_ws;
    _Float16* Xq  = p;            p += NX;
    _Float16* Xk  = p;            p += NX;
    _Float16* Xv  = p;            p += NX;
    _Float16* Wqh = p;            p += NW;
    _Float16* Wkh = p;            p += NW;
    _Float16* Wvh = p;            p += NW;
    _Float16* Woh = p;            p += NW;
    _Float16* Qh  = p;            p += NX;
    _Float16* Kh  = p;            p += NX;
    _Float16* Vt  = p;            p += NX;
    _Float16* AO  = p;            p += NX;

    const int thr = 256;
    cvt_f16_kernel<<<(int)((NX + thr - 1) / thr), thr, 0, stream>>>(query, Xq, (int)NX);
    cvt_f16_kernel<<<(int)((NX + thr - 1) / thr), thr, 0, stream>>>(key,   Xk, (int)NX);
    cvt_f16_kernel<<<(int)((NX + thr - 1) / thr), thr, 0, stream>>>(value, Xv, (int)NX);
    cvt_f16_kernel<<<(int)((NW + thr - 1) / thr), thr, 0, stream>>>(Wq, Wqh, (int)NW);
    cvt_f16_kernel<<<(int)((NW + thr - 1) / thr), thr, 0, stream>>>(Wk, Wkh, (int)NW);
    cvt_f16_kernel<<<(int)((NW + thr - 1) / thr), thr, 0, stream>>>(Wv, Wvh, (int)NW);
    cvt_f16_kernel<<<(int)((NW + thr - 1) / thr), thr, 0, stream>>>(Wo, Woh, (int)NW);

    const int tiles = (BATCH * SEQ / 64) * (DMODEL / 64);   // 2048 waves
    const int gblocks = tiles / 8;                          // 8 waves / block
    gemm_wmma_kernel<<<gblocks, 256, 0, stream>>>(Xq, Wqh, bq, Qh, 0);
    gemm_wmma_kernel<<<gblocks, 256, 0, stream>>>(Xk, Wkh, bk, Kh, 1);
    gemm_wmma_kernel<<<gblocks, 256, 0, stream>>>(Xv, Wvh, bv, Vt, 2);

    const int rope_n = BATCH * NHEAD * SEQ * (HDIM / 2);
    rope_kernel<<<(rope_n + thr - 1) / thr, thr, 0, stream>>>(Qh, Kh);

    attn_kernel<<<dim3(SEQ / 32, BATCH * NHEAD), 32, 0, stream>>>(Qh, Kh, Vt, AO);

    gemm_wmma_kernel<<<gblocks, 256, 0, stream>>>(AO, Woh, bo, out, 3);
}
